// RegionProposalNetwork_52690658787607
// MI455X (gfx1250) — matrix operand order repaired and checked by reading, not compiled, not run on hardware
//
#include <hip/hip_runtime.h>
#include <hip/hip_bf16.h>
#include <math.h>

typedef __attribute__((ext_vector_type(16))) _Float16 v16h;
typedef __attribute__((ext_vector_type(8)))  _Float16 v8h;
typedef __attribute__((ext_vector_type(8)))  float    v8f;
typedef __attribute__((ext_vector_type(4)))  unsigned int u32x4;
typedef __attribute__((ext_vector_type(8)))  int      i32x8;
typedef __attribute__((ext_vector_type(4)))  int      i32x4;

#if defined(__HIP_DEVICE_COMPILE__) && __has_builtin(__builtin_amdgcn_tensor_load_to_lds)
#define USE_TDM 1
#else
#define USE_TDM 0
#endif

#define BM 64
#define BN 64
#define BK 32
#define FEAT 126
#define NA   31752      // 126*126*2 anchors
#define PRE  1000
#define POST 100

__device__ __forceinline__ v8f wmma16(v16h a, v16h b, v8f c) {
  // D = A(16x32 f16) * B(32x16 f16) + C(16x16 f32)
  return __builtin_amdgcn_wmma_f32_16x16x32_f16(false, a, false, b, (short)0, c, false, false);
}

__device__ __forceinline__ v16h cat8(v8h lo, v8h hi) {
  return __builtin_shufflevector(lo, hi, 0,1,2,3,4,5,6,7,8,9,10,11,12,13,14,15);
}

// ---------------------------------------------------------------------------
// NCHW f32 -> NHWC f16 (input image)
// ---------------------------------------------------------------------------
__global__ void nchw_to_nhwc_f16(const float* __restrict__ in, _Float16* __restrict__ out,
                                 int C, int H, int W) {
  size_t i = (size_t)blockIdx.x * blockDim.x + threadIdx.x;
  size_t tot = (size_t)4 * C * H * W;
  if (i >= tot) return;
  // i indexes NHWC output: (((n*H + y)*W + x)*C + c)
  int c = (int)(i % C);
  size_t rem = i / C;
  int x = (int)(rem % W);
  size_t rem2 = rem / W;
  int y = (int)(rem2 % H);
  int n = (int)(rem2 / H);
  out[i] = (_Float16)in[(((size_t)n * C + c) * H + y) * W + x];
}

// ---------------------------------------------------------------------------
// pack conv weights OIHW f32 -> f16 [Cout][K'] with K' = (r*3+s)*Cin + ci
// (tap-major K ordering: each BK=32 K-tile lies under a single (r,s) tap)
// ---------------------------------------------------------------------------
__global__ void pack_w(const float* __restrict__ w, _Float16* __restrict__ pw, int Cout, int Cin) {
  int i = blockIdx.x * blockDim.x + threadIdx.x;
  int tot = Cout * Cin * 9;
  if (i >= tot) return;
  int co  = i / (Cin * 9);
  int rem = i - co * (Cin * 9);
  int ci  = rem / 9;
  int rs  = rem - ci * 9;
  pw[(size_t)co * Cin * 9 + rs * Cin + ci] = (_Float16)w[i];
}

// ---------------------------------------------------------------------------
// implicit-GEMM conv3x3 pad1 (stride 1 or 2), fused bias+BN+ReLU, WMMA f16
// Activations NHWC f16 (in and out).  K ordering: (tap rs, ci): the A gather
// is a single b128 load of 8 consecutive channels at a fixed pixel.
// Weights N-major [Cout][Ktot]; TDM loads the 64(N) x 32(K) tile into LDS
// with padded rows (80B stride).
// grid: (M/64, Cout/64, B)  block: 256 threads (8 wave32)
// ---------------------------------------------------------------------------
template<int STRIDE>
__global__ __launch_bounds__(256)
void conv3x3_wmma(const _Float16* __restrict__ in,   // [B,Hin,Win,Cin] f16
                  const _Float16* __restrict__ wt,   // [Cout][Ktot]    f16 (N-major, tap-major K)
                  const float* __restrict__ bias, const float* __restrict__ gam,
                  const float* __restrict__ bet,  const float* __restrict__ mean,
                  const float* __restrict__ var,
                  _Float16* __restrict__ out,        // [B,Hout,Wout,Cout] f16
                  int Cin, int Cout, int Hin, int Win, int Hout, int Wout)
{
  const int Ktot = Cin * 9;
  const int M    = Hout * Wout;
  const int m0   = blockIdx.x * BM;
  const int n0   = blockIdx.y * BN;
  const int batch = blockIdx.z;
  const int tid  = threadIdx.x;
  const int lane = tid & 31;
  const int wave = tid >> 5;
  const int wm   = wave & 3;   // 4 wave-tiles along M (16 rows each)
  const int wn   = wave >> 2;  // 2 wave-tiles along N (32 cols each)

  __shared__ _Float16 As[BM][BK + 8];   // im2col tile, 80B row stride (16B aligned)
  __shared__ _Float16 Bt[BN][BK + 8];   // weight tile, N-major, 80B row stride

  v8f acc0 = {}; v8f acc1 = {};
  const _Float16* inB = in + (size_t)batch * Hin * Win * Cin;

  // per-thread A-staging coordinates: 8 consecutive ci at one pixel
  const int arow = tid >> 2;
  const int aci  = (tid & 3) * 8;
  const int am   = m0 + arow;
  const int aoy  = am / Wout, aox = am - aoy * Wout;

  for (int rs = 0; rs < 9; ++rs) {
    const int r = rs / 3, s = rs - 3 * r;          // hoisted per-tap math
    const int iy = aoy * STRIDE - 1 + r;
    const int ix = aox * STRIDE - 1 + s;
    const bool ok = ((unsigned)iy < (unsigned)Hin) && ((unsigned)ix < (unsigned)Win);
    const _Float16* srcBase = inB + ((size_t)iy * Win + ix) * Cin + aci;  // NHWC: +c is contiguous

    for (int c0 = 0; c0 < Cin; c0 += BK) {
      const int k0 = rs * Cin + c0;
#if USE_TDM
      // --- TDM: DMA 64(N) x 32(K) f16 weight tile into LDS, padded rows --
      if (tid == 0) {
        unsigned ldsb = (unsigned)(size_t)&Bt[0][0];         // low 32b of generic ptr == LDS offset
        unsigned long long ga = (unsigned long long)(size_t)(wt + (size_t)n0 * Ktot + k0);
        u32x4 g0;
        g0[0] = 1u;                                          // count=1, user-mode D#
        g0[1] = ldsb;                                        // lds_addr
        g0[2] = (unsigned)ga;                                // global_addr[31:0]
        g0[3] = (unsigned)((ga >> 32) & 0x1FFFFFFull) | (2u << 30); // addr[56:32] | type=2
        unsigned td0 = (unsigned)Ktot;                       // tensor_dim0 (K, contiguous)
        unsigned td1 = (unsigned)Cout;                       // tensor_dim1 (N)
        i32x8 g1;
        // data_size=2B | pad_enable | pad_interval=16 DWORDs | pad_amount=4 DWORDs
        g1[0] = (int)((1u << 16) | (1u << 20) | (3u << 22) | (3u << 25));
        g1[1] = (int)((td0 & 0xFFFFu) << 16);                // tensor_dim0[15:0]
        g1[2] = (int)(((td0 >> 16) & 0xFFFFu) | ((td1 & 0xFFFFu) << 16));
        g1[3] = (int)(((td1 >> 16) & 0xFFFFu) | (((unsigned)BK & 0xFFFFu) << 16)); // tile_dim0=32
        g1[4] = (int)((unsigned)BN & 0xFFFFu);               // tile_dim1=64
        g1[5] = (int)(unsigned)Ktot;                         // tensor_dim0_stride low 32
        g1[6] = 0; g1[7] = 0;
        i32x4 g2 = {0,0,0,0}; i32x4 g3 = {0,0,0,0};          // 2-D tensor: groups 2/3 unused
        i32x8 g4 = {0,0,0,0,0,0,0,0};                        // trailing group (unused, zero)
        __builtin_amdgcn_tensor_load_to_lds(g0, g1, g2, g3, g4, 0);
      }
#else
      {   // fallback: vectorized global->LDS weight loads (8 halfs = b128 per thread)
        int e  = tid * 8;
        int nn = e >> 5;                // row (N)
        int kk = e & 31;                // col (K)
        const _Float16* src = wt + (size_t)(n0 + nn) * Ktot + k0 + kk;
        *(uint4*)&Bt[nn][kk] = *(const uint4*)src;
      }
#endif
      // --- im2col gather: one b128 load (8 consecutive channels) ---------
      {
        uint4 v = {0u, 0u, 0u, 0u};
        if (ok) v = *(const uint4*)(srcBase + c0);
        *(uint4*)&As[arow][aci] = v;
      }
      if (lane == 0 && c0 + BK < Cin)
        __builtin_prefetch(srcBase + c0 + BK, 0, 1);
#if USE_TDM
      __builtin_amdgcn_s_wait_tensorcnt(0);
#endif
      __syncthreads();

      // --- fragment fill: all b128 LDS loads (per-lane contiguous) -------
      {
        const int mr   = lane & 15;
        const int half = lane >> 4;
        // A(16x32): lane needs K [16h, 16h+16) contiguous
        const _Float16* ap = &As[wm * 16 + mr][0];
        v16h a = cat8(*(const v8h*)(ap + 16 * half), *(const v8h*)(ap + 16 * half + 8));
        // B(32x16): lane needs column n, K [16h, 16h+16) contiguous in Bt[n][k]
        const _Float16* b0p = &Bt[wn * 32 + mr][16 * half];
        v16h b0 = cat8(*(const v8h*)b0p, *(const v8h*)(b0p + 8));
        const _Float16* b1p = &Bt[wn * 32 + 16 + mr][16 * half];
        v16h b1 = cat8(*(const v8h*)b1p, *(const v8h*)(b1p + 8));
        acc0 = wmma16(a, b0, acc0);
        acc1 = wmma16(a, b1, acc1);
      }
      __syncthreads();
    }
  }

  // --- epilogue: bias + BN + ReLU, store f16 NHWC (coalesced over co) ----
  const int mr   = lane & 15;
  const int half = lane >> 4;
  #pragma unroll
  for (int v = 0; v < 8; ++v) {
    int mg = m0 + wm * 16 + v + 8 * half;
    _Float16* orow = out + ((size_t)batch * M + mg) * Cout;
    {
      int co = n0 + wn * 32 + mr;
      float sc = gam[co] * rsqrtf(var[co] + 1e-5f);
      float y  = (acc0[v] + bias[co] - mean[co]) * sc + bet[co];
      y = y > 0.f ? y : 0.f;
      orow[co] = (_Float16)y;
    }
    {
      int co = n0 + wn * 32 + 16 + mr;
      float sc = gam[co] * rsqrtf(var[co] + 1e-5f);
      float y  = (acc1[v] + bias[co] - mean[co]) * sc + bet[co];
      y = y > 0.f ? y : 0.f;
      orow[co] = (_Float16)y;
    }
  }
}

// ---------------------------------------------------------------------------
// deconv (lhs_dilation) + BN, stride==kernel => exactly one tap per output.
// NHWC input; writes NHWC feature map channels [chanOff, chanOff+256).
// ---------------------------------------------------------------------------
__global__ void deconv_bn(const _Float16* __restrict__ in, const float* __restrict__ w,
    const float* __restrict__ bias, const float* __restrict__ gam, const float* __restrict__ bet,
    const float* __restrict__ mean, const float* __restrict__ var,
    _Float16* __restrict__ feat, int Cin, int Hin, int K, int S, int P, int chanOff)
{
  int pix = blockIdx.x * blockDim.x + threadIdx.x;
  if (pix >= FEAT * FEAT) return;
  int co = blockIdx.y;
  int n  = blockIdx.z;
  int oy = pix / FEAT, ox = pix - oy * FEAT;
  int r = -1, s = -1, iy = 0, ix = 0;
  for (int rr = 0; rr < K; ++rr) { int d = oy + rr - P; if (d >= 0 && d % S == 0 && d / S < Hin) { r = rr; iy = d / S; break; } }
  for (int cc = 0; cc < K; ++cc) { int d = ox + cc - P; if (d >= 0 && d % S == 0 && d / S < Hin) { s = cc; ix = d / S; break; } }
  float acc = 0.f;
  if (r >= 0 && s >= 0) {
    const _Float16* ip = in + (((size_t)n * Hin + iy) * Hin + ix) * Cin;   // NHWC: contiguous ci
    const float* wp = w + (size_t)co * Cin * K * K + r * K + s;
    for (int ci = 0; ci < Cin; ++ci)
      acc += (float)ip[ci] * wp[(size_t)ci * K * K];
  }
  float sc = gam[co] * rsqrtf(var[co] + 1e-5f);
  float y  = (acc + bias[co] - mean[co]) * sc + bet[co];     // deconv path: no ReLU
  feat[((size_t)n * (FEAT * FEAT) + pix) * 768 + chanOff + co] = (_Float16)y;
}

// ---------------------------------------------------------------------------
// 1x1 heads: obj (2) + reg (8), 768 input channels (NHWC: unit-stride reduce);
// emit sigmoid probs and deltas in anchor-major layout (B, y, x, a[, c]).
// ---------------------------------------------------------------------------
__global__ void heads_kernel(const _Float16* __restrict__ feat,
    const float* __restrict__ wc, const float* __restrict__ bc,
    const float* __restrict__ wr, const float* __restrict__ br,
    float* __restrict__ probs, float* __restrict__ deltas)
{
  int pix = blockIdx.x * blockDim.x + threadIdx.x;
  if (pix >= FEAT * FEAT) return;
  int n = blockIdx.y;
  float acc[10];
  #pragma unroll
  for (int i = 0; i < 10; ++i) acc[i] = 0.f;
  const _Float16* f = feat + ((size_t)n * (FEAT * FEAT) + pix) * 768;
  for (int k = 0; k < 768; ++k) {
    float x = (float)f[k];
    acc[0] += x * wc[k];
    acc[1] += x * wc[768 + k];
    #pragma unroll
    for (int c = 0; c < 8; ++c) acc[2 + c] += x * wr[c * 768 + k];
  }
  for (int a = 0; a < 2; ++a) {
    float o = acc[a] + bc[a];
    probs[(size_t)n * NA + pix * 2 + a] = 1.f / (1.f + expf(-o));
    for (int c = 0; c < 4; ++c)
      deltas[((size_t)n * NA + pix * 2 + a) * 4 + c] = acc[2 + a * 4 + c] + br[a * 4 + c];
  }
}

// ---------------------------------------------------------------------------
// anchor decode + clip + validity mask
// ---------------------------------------------------------------------------
__global__ void decode_kernel(const float* __restrict__ deltas, const float* __restrict__ probs,
                              float* __restrict__ boxes, float* __restrict__ msk)
{
  int i = blockIdx.x * blockDim.x + threadIdx.x;
  if (i >= 4 * NA) return;
  int j   = i % NA;
  int y   = j / (FEAT * 2);
  int rem = j - y * (FEAT * 2);
  int x   = rem >> 1;
  int a   = rem & 1;
  float asz = a ? 32.f : 16.f;
  float cxa = (x + 0.5f) * 2.f, cya = (y + 0.5f) * 2.f;
  const float* d = deltas + (size_t)i * 4;
  const float BC = 4.135166556742356f;   // log(1000/16)
  float dw = fminf(d[2], BC), dh = fminf(d[3], BC);
  float cx = d[0] * asz + cxa, cy = d[1] * asz + cya;
  float w = expf(dw) * asz, h = expf(dh) * asz;
  float x1 = fminf(fmaxf(cx - 0.5f * w, 0.f), 256.f);
  float y1 = fminf(fmaxf(cy - 0.5f * h, 0.f), 256.f);
  float x2 = fminf(fmaxf(cx + 0.5f * w, 0.f), 256.f);
  float y2 = fminf(fmaxf(cy + 0.5f * h, 0.f), 256.f);
  float* b = boxes + (size_t)i * 4;
  b[0] = x1; b[1] = y1; b[2] = x2; b[3] = y2;
  float p = probs[i];
  bool valid = (x2 - x1 >= 0.001f) && (y2 - y1 >= 0.001f) && (p >= 0.57f);
  msk[i] = valid ? p : -1.f;
}

// ---------------------------------------------------------------------------
// per-image: top-1000 selection, greedy NMS (IoU>0.7), top-100 emission
// one 1024-thread block per image
// ---------------------------------------------------------------------------
__global__ __launch_bounds__(1024)
void nms_kernel(const float* __restrict__ boxes, float* __restrict__ msk, float* __restrict__ out)
{
  __shared__ float rs[1024];
  __shared__ int   ri[1024];
  __shared__ float sb[PRE][4];
  __shared__ float ss[PRE];
  __shared__ int   keep[PRE];
  const int n = blockIdx.x, tid = threadIdx.x;
  const float* bb = boxes + (size_t)n * NA * 4;
  float* ms = msk + (size_t)n * NA;

  // top-PRE via iterative block argmax (scores consumed in-place)
  for (int i = 0; i < PRE; ++i) {
    float best = -1e30f; int bi = 0;
    for (int j = tid; j < NA; j += 1024) { float v = ms[j]; if (v > best) { best = v; bi = j; } }
    rs[tid] = best; ri[tid] = bi;
    __syncthreads();
    for (int off = 512; off > 0; off >>= 1) {
      if (tid < off && rs[tid + off] > rs[tid]) { rs[tid] = rs[tid + off]; ri[tid] = ri[tid + off]; }
      __syncthreads();
    }
    if (tid == 0) {
      int idx = ri[0]; float sc = rs[0];
      ss[i] = sc;
      sb[i][0] = bb[idx * 4 + 0]; sb[i][1] = bb[idx * 4 + 1];
      sb[i][2] = bb[idx * 4 + 2]; sb[i][3] = bb[idx * 4 + 3];
      keep[i] = sc > 0.f ? 1 : 0;
      ms[idx] = -2e30f;
    }
    __syncthreads();
  }

  // greedy NMS
  for (int i = 0; i < PRE; ++i) {
    __syncthreads();
    if (!keep[i]) continue;
    float ax1 = sb[i][0], ay1 = sb[i][1], ax2 = sb[i][2], ay2 = sb[i][3];
    float aarea = (ax2 - ax1) * (ay2 - ay1);
    for (int j = i + 1 + tid; j < PRE; j += 1024) {
      if (!keep[j]) continue;
      float x1 = fmaxf(ax1, sb[j][0]), y1 = fmaxf(ay1, sb[j][1]);
      float x2 = fminf(ax2, sb[j][2]), y2 = fminf(ay2, sb[j][3]);
      float iw = fmaxf(x2 - x1, 0.f), ih = fmaxf(y2 - y1, 0.f);
      float inter = iw * ih;
      float barea = (sb[j][2] - sb[j][0]) * (sb[j][3] - sb[j][1]);
      float iou = inter / (aarea + barea - inter + 1e-9f);
      if (iou > 0.7f) keep[j] = 0;
    }
  }
  __syncthreads();

  // top-POST among kept; zero rows when exhausted
  for (int k = 0; k < POST; ++k) {
    float best = -1e30f; int bi = 0;
    for (int j = tid; j < PRE; j += 1024) { float v = keep[j] ? ss[j] : -1.f; if (v > best) { best = v; bi = j; } }
    rs[tid] = best; ri[tid] = bi;
    __syncthreads();
    for (int off = 512; off > 0; off >>= 1) {
      if (tid < off && rs[tid + off] > rs[tid]) { rs[tid] = rs[tid + off]; ri[tid] = ri[tid + off]; }
      __syncthreads();
    }
    if (tid == 0) {
      int idx = ri[0]; float sc = rs[0];
      float* o = out + ((size_t)n * POST + k) * 5;
      if (sc > 0.f) {
        o[0] = sb[idx][0]; o[1] = sb[idx][1]; o[2] = sb[idx][2]; o[3] = sb[idx][3]; o[4] = sc;
        keep[idx] = 0;
      } else { o[0] = 0.f; o[1] = 0.f; o[2] = 0.f; o[3] = 0.f; o[4] = 0.f; }
    }
    __syncthreads();
  }
}

// ---------------------------------------------------------------------------
// host launcher
// ---------------------------------------------------------------------------
extern "C" void kernel_launch(void* const* d_in, const int* in_sizes, int n_in,
                              void* d_out, int out_size, void* d_ws, size_t ws_size,
                              hipStream_t stream)
{
  (void)in_sizes; (void)n_in; (void)out_size; (void)ws_size;
  char* ws = (char*)d_ws;
  size_t off = 0;
  auto alloc = [&](size_t bytes) -> char* {
    off = (off + 255) & ~(size_t)255;
    char* p = ws + off; off += bytes; return p;
  };
  _Float16* xh   = (_Float16*)alloc((size_t)4 * 128 * 256 * 256 * 2);   // NHWC
  _Float16* b1a  = (_Float16*)alloc((size_t)4 * 128 * 128 * 128 * 2);
  _Float16* b1b  = (_Float16*)alloc((size_t)4 * 128 * 128 * 128 * 2);
  _Float16* b2a  = (_Float16*)alloc((size_t)4 * 128 * 64 * 64 * 2);
  _Float16* b2b  = (_Float16*)alloc((size_t)4 * 128 * 64 * 64 * 2);
  _Float16* b3a  = (_Float16*)alloc((size_t)4 * 256 * 32 * 32 * 2);
  _Float16* b3b  = (_Float16*)alloc((size_t)4 * 256 * 32 * 32 * 2);
  _Float16* feat = (_Float16*)alloc((size_t)4 * FEAT * FEAT * 768 * 2); // NHWC
  _Float16* pw   = (_Float16*)alloc((size_t)256 * 2304 * 2);            // [Cout][Ktot] f16
  float* probs   = (float*)alloc((size_t)4 * NA * 4);
  float* deltas  = (float*)alloc((size_t)4 * NA * 16);
  float* boxes   = (float*)alloc((size_t)4 * NA * 16);
  float* msk     = (float*)alloc((size_t)4 * NA * 4);

  const float* x = (const float*)d_in[0];
  size_t nx = (size_t)4 * 128 * 256 * 256;
  nchw_to_nhwc_f16<<<(unsigned)((nx + 255) / 256), 256, 0, stream>>>(x, xh, 128, 256, 256);

  auto conv = [&](const _Float16* in, _Float16* outp, int wi, int Cin, int Cout, int Hin, int stride) {
    const float* w    = (const float*)d_in[wi];
    const float* bias = (const float*)d_in[wi + 1];
    const float* g    = (const float*)d_in[wi + 2];
    const float* be   = (const float*)d_in[wi + 3];
    const float* mn   = (const float*)d_in[wi + 4];
    const float* vr   = (const float*)d_in[wi + 5];
    int tot = Cout * Cin * 9;
    pack_w<<<(tot + 255) / 256, 256, 0, stream>>>(w, pw, Cout, Cin);
    int Hout = Hin / stride;
    dim3 grid(Hout * Hout / BM, Cout / BN, 4), blk(256);
    if (stride == 2)
      conv3x3_wmma<2><<<grid, blk, 0, stream>>>(in, pw, bias, g, be, mn, vr, outp, Cin, Cout, Hin, Hin, Hout, Hout);
    else
      conv3x3_wmma<1><<<grid, blk, 0, stream>>>(in, pw, bias, g, be, mn, vr, outp, Cin, Cout, Hin, Hin, Hout, Hout);
  };

  // b1: d_in[1..24], 4 layers, 128->128, 256 -> 128 spatial
  const _Float16* cur = xh;
  _Float16* pp0 = b1a; _Float16* pp1 = b1b;
  for (int i = 0; i < 4; ++i) {
    _Float16* o = (i & 1) ? pp1 : pp0;
    conv(cur, o, 1 + i * 6, 128, 128, (i == 0) ? 256 : 128, (i == 0) ? 2 : 1);
    cur = o;
  }
  const _Float16* b1out = cur;       // b1b
  // b2: d_in[25..60], 6 layers, 128->128, 128 -> 64 spatial
  pp0 = b2a; pp1 = b2b;
  for (int i = 0; i < 6; ++i) {
    _Float16* o = (i & 1) ? pp1 : pp0;
    conv(cur, o, 25 + i * 6, 128, 128, (i == 0) ? 128 : 64, (i == 0) ? 2 : 1);
    cur = o;
  }
  const _Float16* b2out = cur;       // b2b
  // b3: d_in[61..96], 6 layers, 128/256->256, 64 -> 32 spatial
  pp0 = b3a; pp1 = b3b;
  for (int i = 0; i < 6; ++i) {
    _Float16* o = (i & 1) ? pp1 : pp0;
    conv(cur, o, 61 + i * 6, (i == 0) ? 128 : 256, 256, (i == 0) ? 64 : 32, (i == 0) ? 2 : 1);
    cur = o;
  }
  const _Float16* b3out = cur;       // b3b

  dim3 dgrid((FEAT * FEAT + 255) / 256, 256, 4);
  // d1: d_in[97..102]  (256ch, 32 -> 126, k4 s4 p2) -> feat[0:256)
  deconv_bn<<<dgrid, 256, 0, stream>>>(b3out, (const float*)d_in[97], (const float*)d_in[98],
      (const float*)d_in[99], (const float*)d_in[100], (const float*)d_in[101], (const float*)d_in[102],
      feat, 256, 32, 4, 4, 2, 0);
  // d2: d_in[103..108] (128ch, 64 -> 126, k2 s2 p1) -> feat[256:512)
  deconv_bn<<<dgrid, 256, 0, stream>>>(b2out, (const float*)d_in[103], (const float*)d_in[104],
      (const float*)d_in[105], (const float*)d_in[106], (const float*)d_in[107], (const float*)d_in[108],
      feat, 128, 64, 2, 2, 1, 256);
  // d3: d_in[109..114] (128ch, 128 -> 126 crop, k1 s1 p0) -> feat[512:768)
  deconv_bn<<<dgrid, 256, 0, stream>>>(b1out, (const float*)d_in[109], (const float*)d_in[110],
      (const float*)d_in[111], (const float*)d_in[112], (const float*)d_in[113], (const float*)d_in[114],
      feat, 128, 128, 1, 1, 0, 512);

  // heads: cls d_in[115,116], reg d_in[117,118]
  dim3 hgrid((FEAT * FEAT + 255) / 256, 4);
  heads_kernel<<<hgrid, 256, 0, stream>>>(feat, (const float*)d_in[115], (const float*)d_in[116],
      (const float*)d_in[117], (const float*)d_in[118], probs, deltas);

  int tot = 4 * NA;
  decode_kernel<<<(tot + 255) / 256, 256, 0, stream>>>(deltas, probs, boxes, msk);

  nms_kernel<<<4, 1024, 0, stream>>>(boxes, msk, (float*)d_out);
}
